// Attention_57432302682539
// MI455X (gfx1250) — matrix operand order repaired and checked by reading, not compiled, not run on hardware
//
#include <hip/hip_runtime.h>
#include <math.h>

// Problem sizes (compile-time constants from the reference).
constexpr int kB = 32;
constexpr int kS = 4096;
constexpr int kD = 256;
constexpr int kH = 256;

constexpr int kMRows = 64;               // s-rows per workgroup (4 M-tiles per wave)

typedef __attribute__((ext_vector_type(16))) __bf16 v16bf;
typedef __attribute__((ext_vector_type(8)))  float v8f;
typedef __attribute__((ext_vector_type(4)))  int   v4i;

#if __has_builtin(__builtin_amdgcn_global_load_async_to_lds_b128) && \
    __has_builtin(__builtin_amdgcn_s_wait_asynccnt)
#define USE_ASYNC_LDS 1
typedef __attribute__((address_space(1))) v4i GInt4;   // global v4i
typedef __attribute__((address_space(3))) v4i LInt4;   // LDS v4i
#else
#define USE_ASYNC_LDS 0
#endif

__device__ __forceinline__ unsigned short f2bf(float f) {
  union { float f; unsigned u; } c; c.f = f;
  unsigned u = c.u;
  unsigned r = u + 0x7FFFu + ((u >> 16) & 1u);   // round-to-nearest-even
  return (unsigned short)(r >> 16);
}

__device__ __forceinline__ float fast_tanh(float x) {
#if __has_builtin(__builtin_amdgcn_tanhf)
  return __builtin_amdgcn_tanhf(x);
#else
  return tanhf(x);
#endif
}

// ---------------------------------------------------------------------------
// Kernel 1: inpb[br][b][h] = x[b]·W_in[h] + b_in[h] + bc_br[h];  Vcomb[br][h]
// ---------------------------------------------------------------------------
__global__ void prep_kernel(const float* __restrict__ x,
                            const float* __restrict__ W_in,
                            const float* __restrict__ b_in,
                            const float* __restrict__ bc0, const float* __restrict__ bc1,
                            const float* __restrict__ bc2, const float* __restrict__ bc3,
                            const float* __restrict__ V0,  const float* __restrict__ V1,
                            const float* __restrict__ V2,  const float* __restrict__ V3,
                            float* __restrict__ inpb, float* __restrict__ Vcomb) {
  int idx = blockIdx.x * blockDim.x + threadIdx.x;
  if (idx >= 4 * kB * kH) return;
  int h  = idx % kH;
  int b  = (idx / kH) % kB;
  int br = idx / (kB * kH);
  const float* bc = br == 0 ? bc0 : br == 1 ? bc1 : br == 2 ? bc2 : bc3;
  const float* V  = br == 0 ? V0  : br == 1 ? V1  : br == 2 ? V2  : V3;
  float acc = b_in[h] + bc[h];
  const float* xr = x + (size_t)b * kH;
  const float* wr = W_in + (size_t)h * kH;
  for (int k = 0; k < kH; k += 4)
    acc += xr[k] * wr[k] + xr[k+1] * wr[k+1] + xr[k+2] * wr[k+2] + xr[k+3] * wr[k+3];
  inpb[idx] = acc;
  if (b == 0) Vcomb[br * kH + h] = V[h];
}

// ---------------------------------------------------------------------------
// Kernel 2: fp32 Wc -> bf16 copy in workspace (L2-resident, 512 KB total)
// ---------------------------------------------------------------------------
__global__ void cvt_kernel(const float* __restrict__ Wc0, const float* __restrict__ Wc1,
                           const float* __restrict__ Wc2, const float* __restrict__ Wc3,
                           unsigned short* __restrict__ Wcbf) {
  int idx = blockIdx.x * blockDim.x + threadIdx.x;
  if (idx >= 4 * kH * kD) return;
  int br = idx / (kH * kD);
  const float* Wc = br == 0 ? Wc0 : br == 1 ? Wc1 : br == 2 ? Wc2 : Wc3;
  Wcbf[idx] = f2bf(Wc[idx - br * kH * kD]);
}

// ---------------------------------------------------------------------------
// Kernel 3: fused  ctx = context@Wc^T ;  att = V·tanh(inp + ctx) ; 10*tanh
// One workgroup = (b, 64-row s-tile); 4 waves = 4 branches share the LDS tile.
// Each wave computes 4 M-tiles per B-fragment -> 4x Wc reuse from L2.
// ---------------------------------------------------------------------------
__global__ void __launch_bounds__(128)
attn_kernel(const float* __restrict__ context,
            const unsigned char* __restrict__ mask,
            const unsigned short* __restrict__ Wcbf,
            const float* __restrict__ inpb,
            const float* __restrict__ Vcomb,
            float* __restrict__ out) {
  __shared__ __align__(16) unsigned short tile[kMRows * kD];  // 64 x 256 bf16 (32 KB)
#if USE_ASYNC_LDS
  __shared__ __align__(16) float ftile[kMRows * kD];          // fp32 staging (64 KB)
#endif

  const int nsb    = kS / kMRows;
  const int b      = blockIdx.x / nsb;
  const int s_base = (blockIdx.x % nsb) * kMRows;
  const int tid    = threadIdx.x;

  const float* src = context + ((size_t)b * kS + s_base) * kD;

  // ---- Stage context tile into LDS as bf16 -------------------------------
#if USE_ASYNC_LDS
  {
    // 64*256 floats = 4096 16-byte granules; 128 threads x 32 async B128 copies.
    const float4* s4 = (const float4*)src;
    float4* l4 = (float4*)ftile;
    #pragma unroll 4
    for (int i = 0; i < 32; ++i) {
      int g = tid + i * 128;
      __builtin_amdgcn_global_load_async_to_lds_b128(
          (GInt4*)(s4 + g), (LInt4*)(l4 + g), 0, 0);
    }
    __builtin_amdgcn_s_wait_asynccnt(0);
  }
  __syncthreads();
  const float4* csrc = (const float4*)ftile;   // convert out of LDS
#else
  const float4* csrc = (const float4*)src;     // convert straight from global
#endif
  {
    #pragma unroll 4
    for (int i = 0; i < 32; ++i) {
      int e4 = tid + i * 128;                  // float4 granule index
      float4 v = csrc[e4];
      ushort4 o;
      o.x = f2bf(v.x); o.y = f2bf(v.y); o.z = f2bf(v.z); o.w = f2bf(v.w);
      ((ushort4*)tile)[e4] = o;
    }
  }
  __syncthreads();

  const int br     = tid >> 5;       // wave id == branch id
  const int lane   = tid & 31;
  const int lane16 = lane & 15;
  const int half   = lane >> 4;

  const unsigned short* wb     = Wcbf + (size_t)br * kH * kD;
  const float*          inprow = inpb + ((size_t)br * kB + b) * kH;
  const float*          vrow   = Vcomb + br * kH;

  float accOut[4][8];
  #pragma unroll
  for (int mt = 0; mt < 4; ++mt)
    #pragma unroll
    for (int r = 0; r < 8; ++r) accOut[mt][r] = 0.f;

  union Frag { uint4 q[2]; v16bf v; };

  for (int ht = 0; ht < 16; ++ht) {
    const int h = ht * 16 + lane16;                 // N index (output h)
    const unsigned short* bp = wb + (size_t)h * kD; // bf16 row of Wc
    v8f c[4] = {{}, {}, {}, {}};
    #pragma unroll
    for (int k0 = 0; k0 < kD; k0 += 32) {
      // B fragment (32x16 bf16): lanes 0-15 hold K 0..15 of column N=lane,
      // lanes 16-31 hold K 16..31 -> 16 contiguous bf16 from the Wc row.
      Frag Bv;
      const uint4* bq = (const uint4*)(bp + k0 + half * 16);
      Bv.q[0] = bq[0];
      Bv.q[1] = bq[1];
      // 4 A fragments reuse this B fragment (4x L2 traffic reduction).
      #pragma unroll
      for (int mt = 0; mt < 4; ++mt) {
        // A fragment (16x32 bf16): lanes 0-15 hold K 0..15 of row M,
        // lanes 16-31 hold K 16..31 (same contiguous order as B).
        Frag A;
        const int row = (mt * 16 + lane16) * kD;
        A.q[0] = *(const uint4*)&tile[row + k0 + half * 16];
        A.q[1] = *(const uint4*)&tile[row + k0 + half * 16 + 8];
        c[mt] = __builtin_amdgcn_wmma_f32_16x16x32_bf16(false, A.v, false, Bv.v,
                                                        (short)0, c[mt], false, false);
      }
    }
    if (ht < 15) __builtin_prefetch(bp + 16 * kD, 0, 1);  // next h-tile's B rows

    // Epilogue on the C layout: VGPR r, lanes 0-15 -> M=r, lanes 16-31 -> M=r+8,
    // N = lane&15.  acc_s += V[h] * tanh(inp[b,h] + bc[h] + ctx[s,h])
    float ib = inprow[h];
    float vv = vrow[h];
    #pragma unroll
    for (int mt = 0; mt < 4; ++mt)
      #pragma unroll
      for (int r = 0; r < 8; ++r)
        accOut[mt][r] += vv * fast_tanh(ib + c[mt][r]);
  }

  // Reduce over the 16 N-lanes inside each half (masks 1,2,4,8 never cross halves).
  #pragma unroll
  for (int mt = 0; mt < 4; ++mt) {
    #pragma unroll
    for (int r = 0; r < 8; ++r) {
      float a = accOut[mt][r];
      a += __shfl_xor(a, 1);
      a += __shfl_xor(a, 2);
      a += __shfl_xor(a, 4);
      a += __shfl_xor(a, 8);
      accOut[mt][r] = a;
    }
  }

  if (lane16 == 0) {
    #pragma unroll
    for (int mt = 0; mt < 4; ++mt) {
      #pragma unroll
      for (int r = 0; r < 8; ++r) {
        int s = s_base + mt * 16 + half * 8 + r;
        float z = mask[(size_t)b * kS + s] ? 10.0f * fast_tanh(accOut[mt][r]) : -10.0f;
        out[(size_t)b * (4 * kS) + (size_t)br * kS + s] = z;
      }
    }
  }
}

// ---------------------------------------------------------------------------
// Kernel 4: softmax over the BATCH axis (axis=0), in place. One thread/column.
// ---------------------------------------------------------------------------
__global__ void softmax_kernel(float* __restrict__ out) {
  int c = blockIdx.x * blockDim.x + threadIdx.x;
  if (c >= 4 * kS) return;
  const int stride = 4 * kS;
  float m = -3.4e38f;
  for (int b = 0; b < kB; ++b) m = fmaxf(m, out[(size_t)b * stride + c]);
  float sum = 0.f;
  for (int b = 0; b < kB; ++b) {
    float e = __expf(out[(size_t)b * stride + c] - m);
    sum += e;
    out[(size_t)b * stride + c] = e;
  }
  float inv = 1.0f / sum;
  for (int b = 0; b < kB; ++b) out[(size_t)b * stride + c] *= inv;
}

// ---------------------------------------------------------------------------
extern "C" void kernel_launch(void* const* d_in, const int* in_sizes, int n_in,
                              void* d_out, int out_size, void* d_ws, size_t ws_size,
                              hipStream_t stream) {
  const float* x       = (const float*)d_in[0];
  const float* context = (const float*)d_in[1];
  const unsigned char* mask = (const unsigned char*)d_in[2];
  const float* W_in = (const float*)d_in[3];
  const float* b_in = (const float*)d_in[4];
  const float* Wc0  = (const float*)d_in[5];
  const float* bc0  = (const float*)d_in[6];
  const float* Wc1  = (const float*)d_in[7];
  const float* bc1  = (const float*)d_in[8];
  const float* Wc2  = (const float*)d_in[9];
  const float* bc2  = (const float*)d_in[10];
  const float* Wc3  = (const float*)d_in[11];
  const float* bc3  = (const float*)d_in[12];
  const float* V0   = (const float*)d_in[13];
  const float* V1   = (const float*)d_in[14];
  const float* V2   = (const float*)d_in[15];
  const float* V3   = (const float*)d_in[16];

  char* ws = (char*)d_ws;
  unsigned short* Wcbf = (unsigned short*)ws;                                 // 512 KB
  float* inpb  = (float*)(ws + (size_t)4 * kH * kD * sizeof(unsigned short)); // 128 KB
  float* Vcomb = inpb + 4 * kB * kH;                                          // 4 KB
  float* out   = (float*)d_out;

  prep_kernel<<<(4 * kB * kH + 255) / 256, 256, 0, stream>>>(
      x, W_in, b_in, bc0, bc1, bc2, bc3, V0, V1, V2, V3, inpb, Vcomb);
  cvt_kernel<<<(4 * kH * kD + 255) / 256, 256, 0, stream>>>(Wc0, Wc1, Wc2, Wc3, Wcbf);
  attn_kernel<<<kB * (kS / kMRows), 128, 0, stream>>>(context, mask, Wcbf, inpb, Vcomb, out);
  softmax_kernel<<<(4 * kS + 255) / 256, 256, 0, stream>>>(out);
}
